// top_k_gating_5935644803775
// MI455X (gfx1250) — matrix operand order repaired and checked by reading, not compiled, not run on hardware
//
#include <hip/hip_runtime.h>
#include <math.h>

typedef __bf16 bf16;
typedef __bf16 bf16x4 __attribute__((ext_vector_type(4)));
typedef __bf16 bf16x8 __attribute__((ext_vector_type(8)));
typedef __bf16 v16bf  __attribute__((ext_vector_type(16)));
typedef float v8f     __attribute__((ext_vector_type(8)));
typedef float f32x4   __attribute__((ext_vector_type(4)));

#define N_TOK 8192
#define DDIM  4096
#define HDIM  256
#define NEXP  64

// LDS row pads (elements): keep 16B alignment for ds_load_b128, break bank conflicts
#define APAD 72   // 64 + 8  -> 144B row stride
#define BPAD 72
#define WPAD 264  // 256 + 8 -> 528B row stride

static __device__ __forceinline__ v16bf cat8(bf16x8 lo, bf16x8 hi) {
    return __builtin_shufflevector(lo, hi, 0,1,2,3,4,5,6,7,8,9,10,11,12,13,14,15);
}

// ---------------------------------------------------------------------------
// Kernel 1: h = GELU(x @ w1^T + b1), stored as bf16 [N_TOK][HDIM] in d_ws.
// Block = 256 thr (8 waves, 4x2). Block tile 128(M) x 64(H). K staged 64-wide.
// ---------------------------------------------------------------------------
__global__ __launch_bounds__(256, 2)
void gate_gemm_gelu(const float* __restrict__ x, const float* __restrict__ w1,
                    const float* __restrict__ b1, bf16* __restrict__ hbf)
{
    __shared__ __align__(16) bf16 sA[128 * APAD];
    __shared__ __align__(16) bf16 sB[64 * BPAD];

    const int tid      = threadIdx.x;
    const int lane     = tid & 31;
    const int wave     = tid >> 5;
    const int wm       = wave >> 1;      // 0..3  (M)
    const int wn       = wave & 1;       // 0..1  (H)
    const int lane15   = lane & 15;
    const int laneHalf = lane >> 4;      // 0/1 : selects K sub-block per WMMA layout
    const int m0       = blockIdx.x * 128;
    const int h0       = blockIdx.y * 64;

    v8f acc[2][2] = {};

    for (int kb = 0; kb < DDIM; kb += 64) {
        // stage A: 128 rows x 64 K, fp32 -> bf16  (2048 float4 / 256 thr = 8 each)
        #pragma unroll
        for (int i = 0; i < 8; ++i) {
            int f4  = i * 256 + tid;
            int row = f4 >> 4;
            int c4  = f4 & 15;
            f32x4 v = *(const f32x4*)(x + (size_t)(m0 + row) * DDIM + kb + c4 * 4);
            *(bf16x4*)(&sA[row * APAD + c4 * 4]) = __builtin_convertvector(v, bf16x4);
        }
        // stage B: 64 rows (experts-hidden) x 64 K (1024 float4 / 256 thr = 4 each)
        #pragma unroll
        for (int i = 0; i < 4; ++i) {
            int f4  = i * 256 + tid;
            int row = f4 >> 4;
            int c4  = f4 & 15;
            f32x4 v = *(const f32x4*)(w1 + (size_t)(h0 + row) * DDIM + kb + c4 * 4);
            *(bf16x4*)(&sB[row * BPAD + c4 * 4]) = __builtin_convertvector(v, bf16x4);
        }
        __syncthreads();

        #pragma unroll
        for (int ks = 0; ks < 64; ks += 32) {
            v16bf afrag[2], bfrag[2];
            // A 16x32 frag: lanes 0-15 rows M, VGPR0-3 K{0..7}, VGPR4-7 K{16..23};
            // lanes 16-31 same rows, K{8..15}/{24..31}  -> laneHalf*8 element slide.
            #pragma unroll
            for (int mt = 0; mt < 2; ++mt) {
                int row   = wm * 32 + mt * 16 + lane15;
                bf16x8 lo = *(const bf16x8*)(&sA[row * APAD + ks +      laneHalf * 8]);
                bf16x8 hi = *(const bf16x8*)(&sA[row * APAD + ks + 16 + laneHalf * 8]);
                afrag[mt] = cat8(lo, hi);
            }
            // B 32x16 frag: lane = column; lanes 0-15 K{0..7,16..23}, lanes 16-31 K{8..15,24..31}
            #pragma unroll
            for (int nt = 0; nt < 2; ++nt) {
                int col   = wn * 32 + nt * 16 + lane15;
                bf16x8 lo = *(const bf16x8*)(&sB[col * BPAD + ks +      laneHalf * 8]);
                bf16x8 hi = *(const bf16x8*)(&sB[col * BPAD + ks + 16 + laneHalf * 8]);
                bfrag[nt] = cat8(lo, hi);
            }
            #pragma unroll
            for (int mt = 0; mt < 2; ++mt)
                #pragma unroll
                for (int nt = 0; nt < 2; ++nt)
                    acc[mt][nt] = __builtin_amdgcn_wmma_f32_16x16x32_bf16(
                        false, afrag[mt], false, bfrag[nt],
                        (short)0, acc[mt][nt], false, false);
        }
        __syncthreads();
    }

    // epilogue: bias + exact GELU, store bf16 h
    #pragma unroll
    for (int nt = 0; nt < 2; ++nt) {
        int col    = h0 + wn * 32 + nt * 16 + lane15;
        float bias = b1[col];
        #pragma unroll
        for (int mt = 0; mt < 2; ++mt) {
            #pragma unroll
            for (int r = 0; r < 8; ++r) {
                int grow = m0 + wm * 32 + mt * 16 + laneHalf * 8 + r; // C layout: M = r + half*8
                float v  = acc[mt][nt][r] + bias;
                float g  = 0.5f * v * (1.0f + erff(v * 0.70710678118654752f));
                hbf[(size_t)grow * HDIM + col] = (bf16)g;
            }
        }
    }
}

// ---------------------------------------------------------------------------
// Kernel 2: logits = h @ w2^T (WMMA, K=256), softmax, top-2.
// Block = 128 thr (4 waves); 64 tokens/block; wave handles 16 tokens x 64 experts.
// ---------------------------------------------------------------------------
__global__ __launch_bounds__(128, 2)
void router_topk(const bf16* __restrict__ hbf, const float* __restrict__ w2,
                 float* __restrict__ out_idx, float* __restrict__ out_gate,
                 float* __restrict__ out_prob)
{
    __shared__ __align__(16) bf16 sW[NEXP * WPAD];   // w2 as bf16
    __shared__ float sL[64 * 68];                    // logits tile

    const int tid      = threadIdx.x;
    const int lane     = tid & 31;
    const int wave     = tid >> 5;
    const int lane15   = lane & 15;
    const int laneHalf = lane >> 4;
    const int t0       = blockIdx.x * 64;

    // stage w2 (64 x 256 fp32) -> bf16 LDS : 4096 float4 / 128 thr = 32 each
    #pragma unroll
    for (int i = 0; i < 32; ++i) {
        int f4  = i * 128 + tid;
        int row = f4 >> 6;
        int c4  = f4 & 63;
        f32x4 v = *(const f32x4*)(w2 + row * HDIM + c4 * 4);
        *(bf16x4*)(&sW[row * WPAD + c4 * 4]) = __builtin_convertvector(v, bf16x4);
    }
    __syncthreads();

    v8f acc[4] = {};
    const bf16* arow = hbf + (size_t)(t0 + wave * 16 + lane15) * HDIM;

    #pragma unroll
    for (int ks = 0; ks < HDIM; ks += 32) {
        bf16x8 lo = *(const bf16x8*)(arow + ks +      laneHalf * 8);
        bf16x8 hi = *(const bf16x8*)(arow + ks + 16 + laneHalf * 8);
        v16bf a   = cat8(lo, hi);
        #pragma unroll
        for (int nt = 0; nt < 4; ++nt) {
            int col    = nt * 16 + lane15;
            bf16x8 blo = *(const bf16x8*)(&sW[col * WPAD + ks +      laneHalf * 8]);
            bf16x8 bhi = *(const bf16x8*)(&sW[col * WPAD + ks + 16 + laneHalf * 8]);
            acc[nt] = __builtin_amdgcn_wmma_f32_16x16x32_bf16(
                false, a, false, cat8(blo, bhi), (short)0, acc[nt], false, false);
        }
    }

    // scatter logits into LDS: token-major rows of 64 experts
    #pragma unroll
    for (int nt = 0; nt < 4; ++nt)
        #pragma unroll
        for (int r = 0; r < 8; ++r)
            sL[(wave * 16 + laneHalf * 8 + r) * 68 + nt * 16 + lane15] = acc[nt][r];
    __syncthreads();

    // one thread per token: softmax + deterministic top-2 (lowest index wins ties)
    if (tid < 64) {
        const float* row = &sL[tid * 68];
        float m = row[0];
        #pragma unroll 8
        for (int e = 1; e < NEXP; ++e) m = fmaxf(m, row[e]);
        float s = 0.0f;
        #pragma unroll 8
        for (int e = 0; e < NEXP; ++e) s += expf(row[e] - m);
        float inv = 1.0f / s;
        int   tok = t0 + tid;
        float p1 = -1.0f, p2 = -1.0f;
        int   i1 = 0, i2 = 0;
        #pragma unroll 8
        for (int e = 0; e < NEXP; ++e) {
            float p = expf(row[e] - m) * inv;
            out_prob[(size_t)tok * NEXP + e] = p;
            if (p > p1)      { p2 = p1; i2 = i1; p1 = p; i1 = e; }
            else if (p > p2) { p2 = p;  i2 = e; }
        }
        out_idx [tok * 2 + 0] = (float)i1;
        out_idx [tok * 2 + 1] = (float)i2;
        out_gate[tok * 2 + 0] = p1;
        out_gate[tok * 2 + 1] = p2;
    }
}

extern "C" void kernel_launch(void* const* d_in, const int* in_sizes, int n_in,
                              void* d_out, int out_size, void* d_ws, size_t ws_size,
                              hipStream_t stream) {
    const float* x  = (const float*)d_in[0];   // [8192,4096]
    const float* w1 = (const float*)d_in[1];   // [256,4096]
    const float* b1 = (const float*)d_in[2];   // [256]
    const float* w2 = (const float*)d_in[3];   // [64,256]

    bf16*  hbf = (bf16*)d_ws;                  // [8192,256] bf16 = 4 MB
    float* out = (float*)d_out;
    float* out_idx  = out;                     // [8192,2] (indices as float)
    float* out_gate = out + N_TOK * 2;         // [8192,2]
    float* out_prob = out + N_TOK * 4;         // [8192,64]

    dim3 g1(N_TOK / 128, HDIM / 64);           // 64 x 4
    gate_gemm_gelu<<<g1, 256, 0, stream>>>(x, w1, b1, hbf);
    router_topk<<<N_TOK / 64, 128, 0, stream>>>(hbf, w2, out_idx, out_gate, out_prob);
}
